// MultiHeadAttention_86191403696217
// MI455X (gfx1250) — matrix operand order repaired and checked by reading, not compiled, not run on hardware
//
#include <hip/hip_runtime.h>

// Problem dims (fixed by the reference)
#define NB 4
#define NS 2048
#define NC 1024
#define NH 16
#define ND 64
#define NHD (NH * ND)   // 1024

typedef __bf16 bf16_t;
typedef __attribute__((ext_vector_type(4)))  __bf16 v4bf;
typedef __attribute__((ext_vector_type(8)))  __bf16 v8bf;
typedef __attribute__((ext_vector_type(16))) __bf16 v16bf;
typedef __attribute__((ext_vector_type(8)))  float  v8f;

union ABFrag { v16bf v; v8bf h[2]; };

__device__ __forceinline__ bf16_t f2bf(float f) {
  union { float f; unsigned u; } in; in.f = f;
  unsigned u = in.u + 0x7FFFu + ((in.u >> 16) & 1u);   // round-to-nearest-even
  union { unsigned short s; bf16_t b; } out; out.s = (unsigned short)(u >> 16);
  return out.b;
}

// A-fragment (16x32 bf16, MxK). lane&15 = row M; lane>>4 selects K-halves
// {0..7,16..23} vs {8..15,24..31}. Two contiguous 16B loads per lane.
__device__ __forceinline__ v16bf loadA(const bf16_t* base, int ld, int lane) {
  const int row = lane & 15, kh = lane >> 4;
  const bf16_t* p = base + (size_t)row * ld + kh * 8;
  ABFrag f;
  f.h[0] = *reinterpret_cast<const v8bf*>(p);
  f.h[1] = *reinterpret_cast<const v8bf*>(p + 16);
  return f.v;
}

// B-fragment (32x16 bf16, KxN) from a "column-as-row" matrix Bt (row n of Bt
// is column n of B). lane&15 = column N; lane>>4 selects K-halves
// {0..7,16..23} vs {8..15,24..31}. Two contiguous 16B loads per lane.
__device__ __forceinline__ v16bf loadB(const bf16_t* bt, int ld, int lane) {
  const int n = lane & 15, kh = lane >> 4;
  const bf16_t* p = bt + (size_t)n * ld + kh * 8;
  ABFrag f;
  f.h[0] = *reinterpret_cast<const v8bf*>(p);
  f.h[1] = *reinterpret_cast<const v8bf*>(p + 16);
  return f.v;
}

__device__ __forceinline__ v8f wmma_bf16(v16bf a, v16bf b, v8f c) {
  return __builtin_amdgcn_wmma_f32_16x16x32_bf16(false, a, false, b, (short)0, c,
                                                 false, false);
}

// CDNA5 async DMA: copy 16B/lane global -> LDS, tracked by ASYNCcnt (GVS mode).
__device__ __forceinline__ void asyncCopy16(unsigned ldsByteAddr, unsigned gByteOff,
                                            const bf16_t* sbase) {
  asm volatile("global_load_async_to_lds_b128 %0, %1, %2"
               :: "v"(ldsByteAddr), "v"(gByteOff), "s"(sbase) : "memory");
}

// 32x64-tile GEMM helpers: one k=32 step = 2 A-frags + 4 B-frags -> 8 WMMAs.
__device__ __forceinline__ void loadStep(const bf16_t* A, const bf16_t* Bt,
                                         int ldA, int ldB, int kc, int lane,
                                         v16bf a[2], v16bf bf[4]) {
  a[0] = loadA(A + kc, ldA, lane);
  a[1] = loadA(A + (size_t)16 * ldA + kc, ldA, lane);
#pragma unroll
  for (int c = 0; c < 4; ++c)
    bf[c] = loadB(Bt + (size_t)(c * 16) * ldB + kc, ldB, lane);
}

__device__ __forceinline__ void mmaStep(const v16bf a[2], const v16bf bf[4],
                                        v8f acc[2][4]) {
#pragma unroll
  for (int mi = 0; mi < 2; ++mi)
#pragma unroll
    for (int c = 0; c < 4; ++c)
      acc[mi][c] = wmma_bf16(a[mi], bf[c], acc[mi][c]);
}

// ---------------- conversion kernels ----------------

__global__ void k_cvt_f4(const float* __restrict__ x, bf16_t* __restrict__ o, int n4) {
  int i = blockIdx.x * blockDim.x + threadIdx.x;
  if (i >= n4) return;
  float4 f = reinterpret_cast<const float4*>(x)[i];
  v4bf v;
  v[0] = f2bf(f.x); v[1] = f2bf(f.y); v[2] = f2bf(f.z); v[3] = f2bf(f.w);
  reinterpret_cast<v4bf*>(o)[i] = v;
}

// w: [H, C, D] fp32 -> wt: [H, D, C] bf16  (transpose so B-frags are contiguous)
__global__ void k_cvt_wT(const float* __restrict__ w, bf16_t* __restrict__ wt) {
  int i = blockIdx.x * blockDim.x + threadIdx.x;     // over NH*ND*NC
  int h = i >> 16;                                   // ND*NC == 65536
  int rem = i & 65535;
  int d = rem >> 10;                                 // / NC
  int c = rem & 1023;
  wt[i] = f2bf(w[(h << 16) + c * ND + d]);
}

// ----- QKV projection: one wave -> 32x64 tile, copy-free ping-pong pipeline ----

__global__ void __launch_bounds__(32)
k_qkv(const bf16_t* __restrict__ xb,
      const bf16_t* __restrict__ wtq, const bf16_t* __restrict__ wtk,
      const bf16_t* __restrict__ wtv,
      bf16_t* __restrict__ Q, bf16_t* __restrict__ K, bf16_t* __restrict__ Vt) {
  const int lane = threadIdx.x;
  const int qb   = blockIdx.x;            // S/32
  const int h    = blockIdx.y % NH;
  const int mat  = blockIdx.y / NH;       // 0=Q 1=K 2=V
  const int b    = blockIdx.z;

  const bf16_t* A  = xb + (size_t)(b * NS + qb * 32) * NC;
  const bf16_t* wt = (mat == 0) ? wtq : (mat == 1) ? wtk : wtv;
  const bf16_t* Bt = wt + (size_t)h * ND * NC;

  const v8f vzero = {0.f,0.f,0.f,0.f,0.f,0.f,0.f,0.f};
  v8f acc[2][4];
#pragma unroll
  for (int mi = 0; mi < 2; ++mi)
#pragma unroll
    for (int c = 0; c < 4; ++c) acc[mi][c] = vzero;

  // ping-pong fragment sets: no register copies, loads overlap the other set's WMMAs
  v16bf a0[2], b0[4], a1[2], b1[4];
  loadStep(A, Bt, NC, NC, 0,  lane, a0, b0);
  loadStep(A, Bt, NC, NC, 32, lane, a1, b1);

  for (int kc = 0; kc < NC; kc += 64) {
    __builtin_prefetch(A + kc + 256, 0, 3);
    mmaStep(a0, b0, acc);                                   // consume set0 (kc)
    loadStep(A, Bt, NC, NC, (kc + 64) & (NC - 1), lane, a0, b0);
    mmaStep(a1, b1, acc);                                   // consume set1 (kc+32)
    loadStep(A, Bt, NC, NC, (kc + 96) & (NC - 1), lane, a1, b1);
  }

  const int rbase = (lane >> 4) << 3;
  const int col0  = lane & 15;
  if (mat < 2) {                        // Q, K row-major [B,H,S,D]
    bf16_t* out = (mat == 0 ? Q : K) + ((size_t)(b * NH + h) * NS + qb * 32) * ND;
#pragma unroll
    for (int mi = 0; mi < 2; ++mi)
#pragma unroll
      for (int c = 0; c < 4; ++c)
#pragma unroll
        for (int r = 0; r < 8; ++r)
          out[(size_t)(mi * 16 + rbase + r) * ND + c * 16 + col0] =
              f2bf(acc[mi][c][r]);
  } else {                              // V transposed [B,H,D,S] for P*V B-frags
    bf16_t* out = Vt + (size_t)(b * NH + h) * ND * NS;
#pragma unroll
    for (int mi = 0; mi < 2; ++mi)
#pragma unroll
      for (int c = 0; c < 4; ++c)
#pragma unroll
        for (int r = 0; r < 8; ++r)
          out[(size_t)(c * 16 + col0) * NS + qb * 32 + mi * 16 + rbase + r] =
              f2bf(acc[mi][c][r]);
  }
}

// ------- flash attention: 4 waves/block share async-staged K/V tiles -------

#define KVP 72   // padded LDS row pitch (bf16 elems); 144B keeps 16B alignment

__global__ void __launch_bounds__(128)
k_attn(const bf16_t* __restrict__ Q, const bf16_t* __restrict__ K,
       const bf16_t* __restrict__ Vt, bf16_t* __restrict__ O) {
  const int tid  = threadIdx.x;
  const int lane = tid & 31;
  const int w    = tid >> 5;            // wave id 0..3
  const int qb   = blockIdx.x;          // S/64
  const int h    = blockIdx.y;
  const int b    = blockIdx.z;

  __shared__ __attribute__((aligned(16))) bf16_t ldsK[2][64 * KVP];
  __shared__ __attribute__((aligned(16))) bf16_t ldsV[2][64 * KVP];
  __shared__ __attribute__((aligned(16))) bf16_t ldsP[4][16 * KVP];

  const size_t bh = (size_t)(b * NH + h);
  const bf16_t* Kp = K  + bh * NS * ND;             // [S, D]
  const bf16_t* Vp = Vt + bh * ND * NS;             // [D, S]
  const bf16_t* Qp = Q  + (bh * NS + qb * 64 + w * 16) * ND;

  // per-thread async chunk: 4 chunks of 16B each for K and for V per tile
  int cRow[4], cCh[4];
  unsigned kOff[4], vOff[4];
#pragma unroll
  for (int i = 0; i < 4; ++i) {
    int cid = i * 128 + tid;            // 0..511
    cRow[i] = cid >> 3;
    cCh[i]  = cid & 7;
    kOff[i] = (unsigned)((cRow[i] * ND + cCh[i] * 8) * 2);
    vOff[i] = (unsigned)((cRow[i] * NS + cCh[i] * 8) * 2);
  }

  v16bf qa0 = loadA(Qp + 0,  ND, lane);   // loop-invariant Q fragments
  v16bf qa1 = loadA(Qp + 32, ND, lane);

  const v8f vzero = {0.f,0.f,0.f,0.f,0.f,0.f,0.f,0.f};
  v8f oacc[4];
#pragma unroll
  for (int c = 0; c < 4; ++c) oacc[c] = vzero;
  float mrun[8], lrun[8];
#pragma unroll
  for (int r = 0; r < 8; ++r) { mrun[r] = -1e30f; lrun[r] = 0.f; }

  const float sc   = 0.125f * 1.44269504088896f;    // (1/sqrt(D)) * log2(e)
  const int rbase  = (lane >> 4) << 3;
  const int col0   = lane & 15;
  const int rowTop = qb * 64 + w * 16 + rbase;
  const int nTiles = qb + 1;

  // prologue: async-stage tile 0 into buffer 0
#pragma unroll
  for (int i = 0; i < 4; ++i) {
    asyncCopy16((unsigned)(size_t)(&ldsK[0][cRow[i] * KVP + cCh[i] * 8]), kOff[i], Kp);
    asyncCopy16((unsigned)(size_t)(&ldsV[0][cRow[i] * KVP + cCh[i] * 8]), vOff[i], Vp);
  }

  for (int t = 0; t < nTiles; ++t) {
    const int kt  = t * 64;
    const int buf = t & 1;
    if (t + 1 < nTiles) {               // stage next tile into the other buffer
      const bf16_t* kb = Kp + (size_t)(kt + 64) * ND;
      const bf16_t* vb = Vp + (kt + 64);
#pragma unroll
      for (int i = 0; i < 4; ++i) {
        asyncCopy16((unsigned)(size_t)(&ldsK[buf ^ 1][cRow[i] * KVP + cCh[i] * 8]),
                    kOff[i], kb);
        asyncCopy16((unsigned)(size_t)(&ldsV[buf ^ 1][cRow[i] * KVP + cCh[i] * 8]),
                    vOff[i], vb);
      }
      asm volatile("s_wait_asynccnt 0x8" ::: "memory");  // current tile landed
    } else {
      asm volatile("s_wait_asynccnt 0x0" ::: "memory");
    }
    __syncthreads();

    // ---- S = Q * K^T from LDS ----
    v8f s[4];
#pragma unroll
    for (int c = 0; c < 4; ++c) s[c] = vzero;
#pragma unroll
    for (int c = 0; c < 4; ++c) {
      const bf16_t* kb = &ldsK[buf][(c * 16) * KVP];
      v16bf b0 = loadB(kb + 0,  KVP, lane);
      s[c] = wmma_bf16(qa0, b0, s[c]);
      v16bf b1 = loadB(kb + 32, KVP, lane);
      s[c] = wmma_bf16(qa1, b1, s[c]);
    }

    // ---- scale + causal mask (diagonal tiles only) ----
    const bool needMask = (kt + 63 > qb * 64 + w * 16);
#pragma unroll
    for (int c = 0; c < 4; ++c)
#pragma unroll
      for (int r = 0; r < 8; ++r) {
        float v = s[c][r] * sc;
        if (needMask && (kt + c * 16 + col0 > rowTop + r)) v = -1e30f;
        s[c][r] = v;
      }

    // ---- online softmax ----
    float mt[8];
#pragma unroll
    for (int r = 0; r < 8; ++r) {
      float m = fmaxf(fmaxf(s[0][r], s[1][r]), fmaxf(s[2][r], s[3][r]));
#pragma unroll
      for (int msk = 1; msk < 16; msk <<= 1)
        m = fmaxf(m, __shfl_xor(m, msk, 32));
      mt[r] = m;
    }
    float rs[8];
#pragma unroll
    for (int r = 0; r < 8; ++r) {
      float mnew  = fmaxf(mrun[r], mt[r]);
      float alpha = exp2f(mrun[r] - mnew);
      mrun[r] = mnew;
      lrun[r] *= alpha;
#pragma unroll
      for (int c = 0; c < 4; ++c) oacc[c][r] *= alpha;
      float sum = 0.f;
#pragma unroll
      for (int c = 0; c < 4; ++c) {
        float p = exp2f(s[c][r] - mnew);
        s[c][r] = p;
        sum += p;
      }
      rs[r] = sum;
    }
#pragma unroll
    for (int r = 0; r < 8; ++r) {
      float sum = rs[r];
#pragma unroll
      for (int msk = 1; msk < 16; msk <<= 1)
        sum += __shfl_xor(sum, msk, 32);
      lrun[r] += sum;
    }

    // ---- P: accumulator layout -> wave-private LDS -> A-fragments ----
    bf16_t* pT = &ldsP[w][0];
#pragma unroll
    for (int c = 0; c < 4; ++c)
#pragma unroll
      for (int r = 0; r < 8; ++r)
        pT[(rbase + r) * KVP + c * 16 + col0] = f2bf(s[c][r]);
    v16bf pa0 = loadA(pT + 0,  KVP, lane);   // intra-wave LDS is in-order
    v16bf pa1 = loadA(pT + 32, KVP, lane);

    // ---- O += P * V from LDS ----
#pragma unroll
    for (int c = 0; c < 4; ++c) {
      const bf16_t* vb = &ldsV[buf][(c * 16) * KVP];
      v16bf v0 = loadB(vb + 0,  KVP, lane);
      oacc[c] = wmma_bf16(pa0, v0, oacc[c]);
      v16bf v1 = loadB(vb + 32, KVP, lane);
      oacc[c] = wmma_bf16(pa1, v1, oacc[c]);
    }
    __syncthreads();    // everyone done reading buf before it is refilled
  }

  // ---- normalize, emit concat-head output O[B,S,H*D] (bf16) ----
  bf16_t* Op = O + (size_t)(b * NS + qb * 64 + w * 16) * NHD + h * ND;
#pragma unroll
  for (int r = 0; r < 8; ++r) {
    float inv = 1.0f / lrun[r];
#pragma unroll
    for (int c = 0; c < 4; ++c)
      Op[(size_t)(rbase + r) * NHD + c * 16 + col0] = f2bf(oacc[c][r] * inv);
  }
}

// ----- output projection: one wave -> 32x64 tile, copy-free ping-pong -----

__global__ void __launch_bounds__(32)
k_proj(const bf16_t* __restrict__ O, const bf16_t* __restrict__ Wpb,
       const float* __restrict__ bias, float* __restrict__ y) {
  const int lane = threadIdx.x;
  const int mt   = blockIdx.x;          // (B*S)/32
  const int nt   = blockIdx.y;          // C/64
  const bf16_t* A  = O + (size_t)mt * 32 * NHD;
  const bf16_t* Bt = Wpb + (size_t)(nt * 64) * NHD;

  const v8f vzero = {0.f,0.f,0.f,0.f,0.f,0.f,0.f,0.f};
  v8f acc[2][4];
#pragma unroll
  for (int mi = 0; mi < 2; ++mi)
#pragma unroll
    for (int c = 0; c < 4; ++c) acc[mi][c] = vzero;

  v16bf a0[2], b0[4], a1[2], b1[4];
  loadStep(A, Bt, NHD, NHD, 0,  lane, a0, b0);
  loadStep(A, Bt, NHD, NHD, 32, lane, a1, b1);

  for (int kc = 0; kc < NHD; kc += 64) {
    __builtin_prefetch(A + kc + 256, 0, 3);
    mmaStep(a0, b0, acc);
    loadStep(A, Bt, NHD, NHD, (kc + 64) & (NHD - 1), lane, a0, b0);
    mmaStep(a1, b1, acc);
    loadStep(A, Bt, NHD, NHD, (kc + 96) & (NHD - 1), lane, a1, b1);
  }

  const int rbase = (lane >> 4) << 3;
  const int col0  = lane & 15;
#pragma unroll
  for (int c = 0; c < 4; ++c) {
    float bb = bias[nt * 64 + c * 16 + col0];
#pragma unroll
    for (int mi = 0; mi < 2; ++mi)
#pragma unroll
      for (int r = 0; r < 8; ++r)
        y[(size_t)(mt * 32 + mi * 16 + rbase + r) * NC + nt * 64 + c * 16 + col0] =
            acc[mi][c][r] + bb;
  }
}

// ---------------- host orchestration ----------------

extern "C" void kernel_launch(void* const* d_in, const int* in_sizes, int n_in,
                              void* d_out, int out_size, void* d_ws, size_t ws_size,
                              hipStream_t stream) {
  const float* x  = (const float*)d_in[0];
  const float* Wq = (const float*)d_in[1];
  const float* Wk = (const float*)d_in[2];
  const float* Wv = (const float*)d_in[3];
  const float* Wp = (const float*)d_in[4];
  const float* bp = (const float*)d_in[5];
  float* y = (float*)d_out;

  // Workspace layout (bf16 elements). Total ~37.7M elems = ~75.5 MB.
  bf16_t* ws = (bf16_t*)d_ws;
  const size_t nX = (size_t)NB * NS * NC;       // 8,388,608 (== concat-head O size)
  const size_t nW = (size_t)NH * NC * ND;       // 1,048,576
  const size_t nQ = (size_t)NB * NH * NS * ND;  // 8,388,608
  bf16_t* Xb  = ws;            // x in bf16; REUSED as attention output O
  bf16_t* WtQ = Xb  + nX;
  bf16_t* WtK = WtQ + nW;
  bf16_t* WtV = WtK + nW;
  bf16_t* Wpb = WtV + nW;
  bf16_t* Qb  = Wpb + nW;
  bf16_t* Kb  = Qb  + nQ;
  bf16_t* Vtb = Kb  + nQ;

  k_cvt_f4<<<(int)(nX / 4 / 256), 256, 0, stream>>>(x,  Xb,  (int)(nX / 4));
  k_cvt_wT<<<(int)(nW / 256),     256, 0, stream>>>(Wq, WtQ);
  k_cvt_wT<<<(int)(nW / 256),     256, 0, stream>>>(Wk, WtK);
  k_cvt_wT<<<(int)(nW / 256),     256, 0, stream>>>(Wv, WtV);
  k_cvt_f4<<<(int)(nW / 4 / 256), 256, 0, stream>>>(Wp, Wpb, (int)(nW / 4));

  k_qkv <<<dim3(NS / 32, 3 * NH, NB), 32, 0, stream>>>(Xb, WtQ, WtK, WtV, Qb, Kb, Vtb);
  k_attn<<<dim3(NS / 64, NH, NB),    128, 0, stream>>>(Qb, Kb, Vtb, Xb /* O */);
  k_proj<<<dim3((NB * NS) / 32, NC / 64), 32, 0, stream>>>(Xb, Wpb, bp, y);
}
